// CrossAttention_48258252538234
// MI455X (gfx1250) — matrix operand order repaired and checked by reading, not compiled, not run on hardware
//
#include <hip/hip_runtime.h>
#include <hip/hip_bf16.h>

typedef _Float16 h16;
typedef __attribute__((ext_vector_type(16))) _Float16 v16h;
typedef __attribute__((ext_vector_type(8)))  _Float16 v8h;
typedef __attribute__((ext_vector_type(8)))  float    v8f;
typedef __attribute__((ext_vector_type(4)))  float    v4f;
typedef __attribute__((ext_vector_type(4)))  unsigned int v4u;
typedef __attribute__((ext_vector_type(8)))  int      v8i;
typedef __attribute__((ext_vector_type(4)))  int      v4i;

#define B_   2
#define N_   4096
#define D_   512
#define H_   8
#define DH_  64
#define M_   (B_ * N_)          // 8192 rows
#define IN_  (H_ * DH_)         // 512 inner dim
#define SCALE_ 0.125f           // DH^-0.5

#if defined(__has_builtin)
#if __has_builtin(__builtin_amdgcn_tensor_load_to_lds) && \
    __has_builtin(__builtin_amdgcn_s_wait_tensorcnt)
#define HAVE_TDM 1
#endif
#endif

static __device__ __forceinline__ v8f wmma_f16(v16h a, v16h b, v8f c) {
  // D = A(16x32 f16) * B(32x16 f16) + C(16x16 f32)
  return __builtin_amdgcn_wmma_f32_16x16x32_f16(false, a, false, b, (short)0, c,
                                                false, false);
}

static __device__ __forceinline__ v16h concat8(v8h lo, v8h hi) {
  return __builtin_shufflevector(lo, hi, 0, 1, 2, 3, 4, 5, 6, 7, 8, 9, 10, 11,
                                 12, 13, 14, 15);
}

// ---------------------------------------------------------------------------
// Kernel 1: projection  out_f16[bh][n][dh] = (f16)(X[m][:] @ W[:][c])
// block = 128 threads (4 waves); wave w handles rows m0..m0+15, cols n0..n0+63
// ---------------------------------------------------------------------------
__global__ __launch_bounds__(128)
void proj_kernel(const float* __restrict__ X, const float* __restrict__ W,
                 h16* __restrict__ out) {
  __shared__ h16 Wt[64][40];                 // W^T tile, padded rows (80B)

  const int tid = threadIdx.x;
  const int wv = tid >> 5, lane = tid & 31, l15 = lane & 15, g = lane >> 4;
  const int m0 = blockIdx.x * 64 + wv * 16;
  const int n0 = blockIdx.y * 64;
  const int head = n0 >> 6;

  v8f acc[4];
#pragma unroll
  for (int t = 0; t < 4; ++t) acc[t] = {};

  const float* xr = X + (size_t)(m0 + l15) * D_;

  for (int k0 = 0; k0 < D_; k0 += 32) {
    __syncthreads();                         // protect prior-iteration Wt reads
    for (int idx = tid; idx < 2048; idx += 128) {
      int nn = idx & 63, kk = idx >> 6;
      Wt[nn][kk] = (h16)W[(size_t)(k0 + kk) * IN_ + (n0 + nn)];
    }
    __syncthreads();

    // A fragment: lane = row l15; elems 0..7 -> K=k0+8g+i, 8..15 -> K=k0+16+8g+i
    const float* p = xr + k0 + 8 * g;
    v4f u0 = *(const v4f*)(p);
    v4f u1 = *(const v4f*)(p + 4);
    v4f u2 = *(const v4f*)(p + 16);
    v4f u3 = *(const v4f*)(p + 20);
    v16h a;
#pragma unroll
    for (int i = 0; i < 4; ++i) {
      a[i] = (h16)u0[i]; a[4 + i] = (h16)u1[i];
      a[8 + i] = (h16)u2[i]; a[12 + i] = (h16)u3[i];
    }
#pragma unroll
    for (int t = 0; t < 4; ++t) {
      const v8h* wr = (const v8h*)&Wt[16 * t + l15][0];
      acc[t] = wmma_f16(a, concat8(wr[g], wr[2 + g]), acc[t]);
    }
  }

#pragma unroll
  for (int t = 0; t < 4; ++t) {
#pragma unroll
    for (int r = 0; r < 8; ++r) {
      int m = m0 + r + 8 * g;
      int b_idx = m >> 12;
      int n_idx = m & (N_ - 1);
      out[(((size_t)(b_idx * H_ + head)) * N_ + n_idx) * DH_ + 16 * t + l15] =
          (h16)acc[t][r];
    }
  }
}

// ---------------------------------------------------------------------------
// TDM descriptor: 2D tile (32 keys x 64 dh, f16) from global -> LDS
// ---------------------------------------------------------------------------
#ifdef HAVE_TDM
static __device__ __forceinline__ void tdm_load_chunk(const h16* gsrc,
                                                      h16* ldst) {
  unsigned long long ga = (unsigned long long)(uintptr_t)gsrc;
  unsigned int lds = (unsigned int)(uintptr_t)ldst;  // flat->LDS: low 32 bits
  v4u g0;
  g0[0] = 1u;                                        // count=1, user mode
  g0[1] = lds;                                       // lds_addr
  g0[2] = (unsigned int)ga;                          // global_addr[31:0]
  g0[3] = (unsigned int)((ga >> 32) & 0x01FFFFFFu) | 0x80000000u;  // type=2
  v8i g1;
  g1[0] = 0x00010000;                 // workgroup_mask=0, data_size=1 (2B)
  g1[1] = (int)(64u << 16);           // tensor_dim0 = 64 (lo16)
  g1[2] = (int)(4096u << 16);         // tensor_dim0 hi=0, tensor_dim1 = 4096
  g1[3] = (int)(64u << 16);           // tensor_dim1 hi=0, tile_dim0 = 64
  g1[4] = 32;                         // tile_dim1 = 32, tile_dim2 = 0
  g1[5] = 64;                         // tensor_dim0_stride = 64
  g1[6] = 0;
  g1[7] = 0;
  v4i z4 = {};
  v8i z8 = {};
  __builtin_amdgcn_tensor_load_to_lds(g0, g1, z4, z4, z8, 0);
}
#endif

// ---------------------------------------------------------------------------
// Kernel 2: flash attention per (b,h).  grid = (N/64, B*H), block = 128 (4 waves)
// K chunks: TDM async load, double-buffered.  V chunks: manual transpose stage.
// ---------------------------------------------------------------------------
__global__ __launch_bounds__(128)
void attn_kernel(const h16* __restrict__ qws, const h16* __restrict__ kws,
                 const h16* __restrict__ vws, h16* __restrict__ ows) {
  __shared__ h16 Klds[2][32][64];            // double-buffered [key_local][dh]
  __shared__ h16 Vt[64][40];                 // [dh][key_local], padded
  __shared__ h16 Plds[4][16][32];            // per-wave P staging

  const int tid = threadIdx.x;
  const int wv = tid >> 5, lane = tid & 31, l15 = lane & 15, g = lane >> 4;
  const int bh = blockIdx.y;
  const int qm0 = blockIdx.x * 64 + wv * 16;

  // Q fragments (dh split in two K=32 steps), loaded once
  const h16* qrow = qws + (((size_t)bh) * N_ + qm0 + l15) * DH_;
  v16h qf[2];
#pragma unroll
  for (int s = 0; s < 2; ++s) {
    const v8h* qp = (const v8h*)qrow;
    qf[s] = concat8(qp[4 * s + g], qp[4 * s + 2 + g]);
  }

  v8f o[4];
#pragma unroll
  for (int t = 0; t < 4; ++t) o[t] = {};
  float mstate[8], lstate[8];
#pragma unroll
  for (int r = 0; r < 8; ++r) { mstate[r] = -1e30f; lstate[r] = 0.0f; }

  const h16* kbase = kws + ((size_t)bh) * N_ * DH_;
  const h16* vbase = vws + ((size_t)bh) * N_ * DH_;

#ifdef HAVE_TDM
  if (wv == 0) tdm_load_chunk(kbase, &Klds[0][0][0]);   // prefetch chunk 0
#endif

  for (int jc = 0; jc < N_; jc += 32) {
    const int cur = (jc >> 5) & 1;
    __syncthreads();   // all waves done with previous chunk (frees buf cur^1)

#ifndef HAVE_TDM
    {                                        // fallback: synchronous K copy
      const v8h* src = (const v8h*)(kbase + (size_t)jc * DH_);
      v8h* dst = (v8h*)&Klds[cur][0][0];
#pragma unroll
      for (int i = 0; i < 2; ++i) dst[tid + 128 * i] = src[tid + 128 * i];
    }
#endif
    {                                        // stage V chunk transposed
#pragma unroll
      for (int i = 0; i < 2; ++i) {
        int lin = tid + 128 * i;             // 0..255
        int key = lin >> 3;
        int dh0 = (lin & 7) * 8;
        v8h vsrc = *(const v8h*)(vbase + ((size_t)(jc + key)) * DH_ + dh0);
#pragma unroll
        for (int j = 0; j < 8; ++j) Vt[dh0 + j][key] = vsrc[j];
      }
    }
#ifdef HAVE_TDM
    if (wv == 0) {
      if (jc + 32 < N_) {                    // issue next chunk, retire current
        tdm_load_chunk(kbase + (size_t)(jc + 32) * DH_, &Klds[cur ^ 1][0][0]);
        __builtin_amdgcn_s_wait_tensorcnt(1);
      } else {
        __builtin_amdgcn_s_wait_tensorcnt(0);
      }
    }
#endif
    __syncthreads();

    // ---- S = Q * K^T  (16 x 32 scores as two 16x16 tiles) ----
    v8f s0 = {}, s1 = {};
#pragma unroll
    for (int s = 0; s < 2; ++s) {
      const v8h* kr0 = (const v8h*)&Klds[cur][l15][0];
      const v8h* kr1 = (const v8h*)&Klds[cur][16 + l15][0];
      s0 = wmma_f16(qf[s], concat8(kr0[4 * s + g], kr0[4 * s + 2 + g]),
                    s0);
      s1 = wmma_f16(qf[s], concat8(kr1[4 * s + g], kr1[4 * s + 2 + g]),
                    s1);
    }

    // ---- online softmax over the 32-key chunk ----
#pragma unroll
    for (int r = 0; r < 8; ++r) {
      float va = s0[r] * SCALE_, vb = s1[r] * SCALE_;
      float mx = fmaxf(va, vb);
#pragma unroll
      for (int off = 1; off < 16; off <<= 1)
        mx = fmaxf(mx, __shfl_xor(mx, off, 32));
      float mnew = fmaxf(mstate[r], mx);
      float corr = __expf(mstate[r] - mnew);
      float p0 = __expf(va - mnew);
      float p1 = __expf(vb - mnew);
      float rs = p0 + p1;
#pragma unroll
      for (int off = 1; off < 16; off <<= 1) rs += __shfl_xor(rs, off, 32);
      lstate[r] = lstate[r] * corr + rs;
      mstate[r] = mnew;
#pragma unroll
      for (int t = 0; t < 4; ++t) o[t][r] *= corr;
      Plds[wv][r + 8 * g][l15] = (h16)p0;
      Plds[wv][r + 8 * g][16 + l15] = (h16)p1;
    }

    // ---- P fragment (C-layout -> A-layout via per-wave LDS) ----
    const v8h* pr = (const v8h*)&Plds[wv][l15][0];
    v16h ap = concat8(pr[g], pr[2 + g]);

    // ---- O += P * V ----
#pragma unroll
    for (int t = 0; t < 4; ++t) {
      const v8h* vr = (const v8h*)&Vt[16 * t + l15][0];
      o[t] = wmma_f16(ap, concat8(vr[g], vr[2 + g]), o[t]);
    }
  }

  // epilogue: normalize and store to [b][n][h*64+dh] f16
  const int b_idx = bh >> 3, head = bh & 7;
#pragma unroll
  for (int t = 0; t < 4; ++t) {
#pragma unroll
    for (int r = 0; r < 8; ++r) {
      int n_idx = qm0 + r + 8 * g;
      float val = o[t][r] / lstate[r];
      ows[((size_t)b_idx * N_ + n_idx) * IN_ + head * DH_ + 16 * t + l15] =
          (h16)val;
    }
  }
}

// ---------------------------------------------------------------------------
// Kernel 3: output projection  out_f32[m][c] = o_f16[m][:] @ Wo[:][c] + bo[c]
// ---------------------------------------------------------------------------
__global__ __launch_bounds__(128)
void outproj_kernel(const h16* __restrict__ A, const float* __restrict__ Wo,
                    const float* __restrict__ bo, float* __restrict__ out) {
  __shared__ h16 Wt[64][40];

  const int tid = threadIdx.x;
  const int wv = tid >> 5, lane = tid & 31, l15 = lane & 15, g = lane >> 4;
  const int m0 = blockIdx.x * 64 + wv * 16;
  const int n0 = blockIdx.y * 64;

  v8f acc[4];
#pragma unroll
  for (int t = 0; t < 4; ++t) acc[t] = {};

  const h16* ar = A + (size_t)(m0 + l15) * IN_;

  for (int k0 = 0; k0 < IN_; k0 += 32) {
    __syncthreads();
    for (int idx = tid; idx < 2048; idx += 128) {
      int nn = idx & 63, kk = idx >> 6;
      Wt[nn][kk] = (h16)Wo[(size_t)(k0 + kk) * D_ + (n0 + nn)];
    }
    __syncthreads();

    const v8h* ap = (const v8h*)ar;
    v16h a = concat8(ap[(k0 >> 3) + g], ap[(k0 >> 3) + 2 + g]);
#pragma unroll
    for (int t = 0; t < 4; ++t) {
      const v8h* wr = (const v8h*)&Wt[16 * t + l15][0];
      acc[t] = wmma_f16(a, concat8(wr[g], wr[2 + g]), acc[t]);
    }
  }

#pragma unroll
  for (int t = 0; t < 4; ++t) {
#pragma unroll
    for (int r = 0; r < 8; ++r) {
      int m = m0 + r + 8 * g;
      int c = n0 + 16 * t + l15;
      out[(size_t)m * D_ + c] = acc[t][r] + bo[c];
    }
  }
}

// ---------------------------------------------------------------------------
extern "C" void kernel_launch(void* const* d_in, const int* in_sizes, int n_in,
                              void* d_out, int out_size, void* d_ws,
                              size_t ws_size, hipStream_t stream) {
  (void)in_sizes; (void)n_in; (void)out_size; (void)ws_size;
  const float* x  = (const float*)d_in[0];
  const float* Wq = (const float*)d_in[1];
  const float* Wk = (const float*)d_in[2];
  const float* Wv = (const float*)d_in[3];
  const float* Wo = (const float*)d_in[4];
  const float* bo = (const float*)d_in[5];
  float* out = (float*)d_out;

  const size_t slab = (size_t)B_ * H_ * N_ * DH_;  // 4M f16 elements = 8 MB
  h16* qws = (h16*)d_ws;
  h16* kws = qws + slab;
  h16* vws = kws + slab;
  h16* ows = vws + slab;                            // total 32 MB workspace

  dim3 blk(128);
  dim3 gproj(M_ / 64, IN_ / 64);                    // 128 x 8
  proj_kernel<<<gproj, blk, 0, stream>>>(x, Wq, qws);
  proj_kernel<<<gproj, blk, 0, stream>>>(x, Wk, kws);
  proj_kernel<<<gproj, blk, 0, stream>>>(x, Wv, vws);

  dim3 gattn(N_ / 64, B_ * H_);                     // 64 x 16
  attn_kernel<<<gattn, blk, 0, stream>>>(qws, kws, vws, ows);

  dim3 gout(M_ / 64, D_ / 64);                      // 128 x 8
  outproj_kernel<<<gout, blk, 0, stream>>>(ows, Wo, bo, out);
}
